// CausalMultiHeadSelfAttention_17824114278944
// MI455X (gfx1250) — matrix operand order repaired and checked by reading, not compiled, not run on hardware
//
#include <hip/hip_runtime.h>
#include <hip/hip_bf16.h>

#define D_MODEL 1024
#define N_HEADS 16
#define DK      64
#define SEQ     4096
#define BATCH   2
#define MTOT    (BATCH * SEQ)

typedef __attribute__((ext_vector_type(16))) __bf16 v16bf;
typedef __attribute__((ext_vector_type(8)))  __bf16 v8bf;
typedef __attribute__((ext_vector_type(8)))  float  v8f;

__device__ __forceinline__ v16bf cat8(v8bf lo, v8bf hi) {
  return __builtin_shufflevector(lo, hi, 0, 1, 2, 3, 4, 5, 6, 7,
                                         8, 9, 10, 11, 12, 13, 14, 15);
}
__device__ __forceinline__ v16bf ld32(const __bf16* p) {
  const v8bf* q = (const v8bf*)p;
  return cat8(q[0], q[1]);
}
__device__ __forceinline__ v8f zero8() {
  v8f z;
#pragma unroll
  for (int e = 0; e < 8; ++e) z[e] = 0.0f;
  return z;
}
__device__ __forceinline__ v8f wmma_bf(v16bf a, v16bf b, v8f c) {
  return __builtin_amdgcn_wmma_f32_16x16x32_bf16(false, a, false, b,
                                                 (short)0, c, false, false);
}

// ---- DPP row-rotate (within 16-lane rows) reductions on the VALU pipe ------
// DPP ctrl: ROW_ROR = 0x120 + n
template <int N>
__device__ __forceinline__ float dpp_ror(float x) {
  int s = __float_as_int(x);
  int r = __builtin_amdgcn_update_dpp(s, s, 0x120 + N, 0xF, 0xF, false);
  return __int_as_float(r);
}
__device__ __forceinline__ float rowmax16(float x) {
  x = fmaxf(x, dpp_ror<8>(x));
  x = fmaxf(x, dpp_ror<4>(x));
  x = fmaxf(x, dpp_ror<2>(x));
  x = fmaxf(x, dpp_ror<1>(x));
  return x;
}
__device__ __forceinline__ float rowsum16(float x) {
  x += dpp_ror<8>(x);
  x += dpp_ror<4>(x);
  x += dpp_ror<2>(x);
  x += dpp_ror<1>(x);
  return x;
}

// ---------------------------------------------------------------- f32 -> bf16
__global__ void cvt_bf16_kernel(const float* __restrict__ src,
                                __bf16* __restrict__ dst, int n8) {
  int i = blockIdx.x * blockDim.x + threadIdx.x;
  if (i >= n8) return;
  const float4* s = (const float4*)src + (size_t)i * 2;
  float4 a = s[0], b = s[1];
  v8bf o;
  o[0] = (__bf16)a.x; o[1] = (__bf16)a.y; o[2] = (__bf16)a.z; o[3] = (__bf16)a.w;
  o[4] = (__bf16)b.x; o[5] = (__bf16)b.y; o[6] = (__bf16)b.z; o[7] = (__bf16)b.w;
  ((v8bf*)dst)[i] = o;
}

// ------------------------------------------------------------------ GEMM core
// out[m, o] = sum_d A[m, d] * W[o, d];  M = 8192, N = K = 1024.
// Each wave: 16 (M) x 128 (N) strip -> 8 accumulators, A-fragment reused 8x.
// MODE 0: store bf16 into [B, H, S, DK]        (Q, K projections)
// MODE 1: store bf16 into [B, H, DK, S] (V^T)  (V projection)
// MODE 2: store f32  into [M, N]               (output projection)
template <int MODE>
__global__ void __launch_bounds__(128)
gemm_wmma_kernel(const __bf16* __restrict__ A, const __bf16* __restrict__ W,
                 __bf16* __restrict__ outb, float* __restrict__ outf) {
  const int lane = threadIdx.x & 31;
  const int wave = threadIdx.x >> 5;
  const int col  = lane & 15;
  const int hi   = lane >> 4;
  const int mTile = blockIdx.x * 64 + wave * 16;
  const int nTile = blockIdx.y * 128;

  v8f acc[8];
#pragma unroll
  for (int j = 0; j < 8; ++j) acc[j] = zero8();

  const __bf16* arow = A + (size_t)(mTile + col) * D_MODEL;
  for (int kk = 0; kk < D_MODEL; kk += 32) {
    const int koff = kk + hi * 16;
    v16bf af = cat8(*(const v8bf*)(arow + koff),
                    *(const v8bf*)(arow + koff + 8));
#pragma unroll
    for (int j = 0; j < 8; ++j) {
      const __bf16* bp =
          W + (size_t)(nTile + j * 16 + col) * D_MODEL + kk + hi * 16;
      acc[j] = wmma_bf(af, ld32(bp), acc[j]);
    }
  }

  if (MODE == 0) {
#pragma unroll
    for (int j = 0; j < 8; ++j) {
      int o = nTile + j * 16 + col;
      int h = o >> 6, dk = o & 63;
#pragma unroll
      for (int r = 0; r < 8; ++r) {
        int m = mTile + hi * 8 + r;
        int b = m >> 12, s = m & (SEQ - 1);
        outb[((size_t)(b * N_HEADS + h) * SEQ + s) * DK + dk] =
            (__bf16)acc[j][r];
      }
    }
  } else if (MODE == 1) {
#pragma unroll
    for (int j = 0; j < 8; ++j) {
      int o = nTile + j * 16 + col;
      int h = o >> 6, dk = o & 63;
      int m0 = mTile + hi * 8;
      int b = m0 >> 12, s0 = m0 & (SEQ - 1);
      v8bf pk;
#pragma unroll
      for (int r = 0; r < 8; ++r) pk[r] = (__bf16)acc[j][r];
      *(v8bf*)(outb + ((size_t)(b * N_HEADS + h) * DK + dk) * SEQ + s0) = pk;
    }
  } else {
#pragma unroll
    for (int j = 0; j < 8; ++j) {
      int o = nTile + j * 16 + col;
#pragma unroll
      for (int r = 0; r < 8; ++r) {
        int m = mTile + hi * 8 + r;
        outf[(size_t)m * D_MODEL + o] = acc[j][r];
      }
    }
  }
}

// ----------------------------------------------------------------------- RoPE
// outScale folds the 1/sqrt(d_k) attention scale into Q (K uses 1.0).
__global__ void rope_kernel(__bf16* __restrict__ buf,
                            const int* __restrict__ pos, float outScale,
                            int npairs) {
  int p = blockIdx.x * blockDim.x + threadIdx.x;
  if (p >= npairs) return;
  int i  = p & 31;
  int s  = (p >> 5) & (SEQ - 1);
  int bh = p >> 17;
  // theta^(-i/32) = 2^(-i * log2(10000)/32)
  float inv = exp2f(-(float)i * 0.41524101186092057f);
  float fr = (float)pos[s] * inv;
  float sn, cs;
  __sincosf(fr, &sn, &cs);
  size_t base = ((size_t)bh * SEQ + s) * DK + 2 * i;
  float xe = (float)buf[base];
  float xo = (float)buf[base + 1];
  buf[base]     = (__bf16)((xe * cs - xo * sn) * outScale);
  buf[base + 1] = (__bf16)((xe * sn + xo * cs) * outScale);
}

// ------------------------------------------------------ causal flash attention
// Q: [B*H, S, DK] bf16 (RoPE + 1/sqrt(dk) applied); K: [B*H, S, DK] bf16;
// Vt: [B*H, DK, S] bf16;  O: [B, S, D_MODEL] bf16
__global__ void __launch_bounds__(128)
flash_attn_kernel(const __bf16* __restrict__ Q, const __bf16* __restrict__ K,
                  const __bf16* __restrict__ Vt, __bf16* __restrict__ O) {
  __shared__ __bf16 psh[4][16 * 32];
  const int lane = threadIdx.x & 31;
  const int wave = threadIdx.x >> 5;
  const int col  = lane & 15;
  const int hi   = lane >> 4;
  const int bh   = blockIdx.y;
  const int qTile = blockIdx.x * 64 + wave * 16;

  const int koff = hi * 16;
  const __bf16* qrow = Q + ((size_t)bh * SEQ + qTile + col) * DK;
  v16bf qa0 = cat8(*(const v8bf*)(qrow + koff),
                   *(const v8bf*)(qrow + koff + 8));
  v16bf qa1 = cat8(*(const v8bf*)(qrow + 32 + koff),
                   *(const v8bf*)(qrow + 32 + koff + 8));

  const __bf16* kbh = K + (size_t)bh * SEQ * DK;
  const __bf16* vbh = Vt + (size_t)bh * DK * SEQ;

  v8f oacc[4];
#pragma unroll
  for (int j = 0; j < 4; ++j) oacc[j] = zero8();
  float mrow[8], lrow[8];
#pragma unroll
  for (int r = 0; r < 8; ++r) { mrow[r] = -1e30f; lrow[r] = 0.0f; }

  __bf16* pb = &psh[wave][0];
  const int qbase = qTile + hi * 8;

  int kb = 0;
  // ---------------- interior blocks: every key strictly below the diagonal
  for (; kb + 32 <= qTile; kb += 32) {
    v8f c0 = zero8(), c1 = zero8();
    const __bf16* k0 = kbh + (size_t)(kb + col) * DK + hi * 16;
    c0 = wmma_bf(qa0, ld32(k0), c0);
    c0 = wmma_bf(qa1, ld32(k0 + 32), c0);
    const __bf16* k1 = kbh + (size_t)(kb + 16 + col) * DK + hi * 16;
    c1 = wmma_bf(qa0, ld32(k1), c1);
    c1 = wmma_bf(qa1, ld32(k1 + 32), c1);
    // prefetch next K / V^T blocks into near cache while softmax runs
    __builtin_prefetch((const void*)(kbh + (size_t)(kb + 32 + col) * DK), 0, 3);
    __builtin_prefetch((const void*)(vbh + (size_t)col * SEQ + kb + 32), 0, 3);

    float alpha[8];
#pragma unroll
    for (int r = 0; r < 8; ++r) {
      float tm = rowmax16(fmaxf(c0[r], c1[r]));
      float mn = fmaxf(mrow[r], tm);
      alpha[r] = __expf(mrow[r] - mn);
      mrow[r] = mn;
      float p0 = __expf(c0[r] - mn);
      float p1 = __expf(c1[r] - mn);
      c0[r] = p0; c1[r] = p1;
      lrow[r] = lrow[r] * alpha[r] + rowsum16(p0 + p1);
#pragma unroll
      for (int j = 0; j < 4; ++j) oacc[j][r] = oacc[j][r] * alpha[r];
    }

#pragma unroll
    for (int r = 0; r < 8; ++r) {
      pb[(hi * 8 + r) * 32 + col]      = (__bf16)c0[r];
      pb[(hi * 8 + r) * 32 + 16 + col] = (__bf16)c1[r];
    }
    v16bf pa = cat8(*(const v8bf*)(pb + col * 32 + koff),
                    *(const v8bf*)(pb + col * 32 + koff + 8));
#pragma unroll
    for (int j = 0; j < 4; ++j) {
      const __bf16* vp = vbh + (size_t)(j * 16 + col) * SEQ + kb + hi * 16;
      oacc[j] = wmma_bf(pa, ld32(vp), oacc[j]);
    }
  }

  // ---------------- diagonal block(s): apply the causal mask
  for (; kb < qTile + 16; kb += 32) {
    v8f c0 = zero8(), c1 = zero8();
    const __bf16* k0 = kbh + (size_t)(kb + col) * DK + hi * 16;
    c0 = wmma_bf(qa0, ld32(k0), c0);
    c0 = wmma_bf(qa1, ld32(k0 + 32), c0);
    const __bf16* k1 = kbh + (size_t)(kb + 16 + col) * DK + hi * 16;
    c1 = wmma_bf(qa0, ld32(k1), c1);
    c1 = wmma_bf(qa1, ld32(k1 + 32), c1);

    const int kc0 = kb + col, kc1 = kb + 16 + col;
    float alpha[8];
#pragma unroll
    for (int r = 0; r < 8; ++r) {
      int q = qbase + r;
      float s0 = (kc0 <= q) ? c0[r] : -1e30f;
      float s1 = (kc1 <= q) ? c1[r] : -1e30f;
      float tm = rowmax16(fmaxf(s0, s1));
      float mn = fmaxf(mrow[r], tm);
      alpha[r] = __expf(mrow[r] - mn);
      mrow[r] = mn;
      float p0 = __expf(s0 - mn);  // masked -1e30 underflows to exactly 0
      float p1 = __expf(s1 - mn);
      c0[r] = p0; c1[r] = p1;
      lrow[r] = lrow[r] * alpha[r] + rowsum16(p0 + p1);
#pragma unroll
      for (int j = 0; j < 4; ++j) oacc[j][r] = oacc[j][r] * alpha[r];
    }

#pragma unroll
    for (int r = 0; r < 8; ++r) {
      pb[(hi * 8 + r) * 32 + col]      = (__bf16)c0[r];
      pb[(hi * 8 + r) * 32 + 16 + col] = (__bf16)c1[r];
    }
    v16bf pa = cat8(*(const v8bf*)(pb + col * 32 + koff),
                    *(const v8bf*)(pb + col * 32 + koff + 8));
#pragma unroll
    for (int j = 0; j < 4; ++j) {
      const __bf16* vp = vbh + (size_t)(j * 16 + col) * SEQ + kb + hi * 16;
      oacc[j] = wmma_bf(pa, ld32(vp), oacc[j]);
    }
  }

  float rinv[8];
#pragma unroll
  for (int r = 0; r < 8; ++r) rinv[r] = 1.0f / lrow[r];

  const int b = bh >> 4, h = bh & 15;
#pragma unroll
  for (int j = 0; j < 4; ++j)
#pragma unroll
    for (int r = 0; r < 8; ++r) {
      int s = qTile + hi * 8 + r;
      O[(size_t)(b * SEQ + s) * D_MODEL + h * 64 + j * 16 + col] =
          (__bf16)(oacc[j][r] * rinv[r]);
    }
}

// -------------------------------------------------------------------- launcher
extern "C" void kernel_launch(void* const* d_in, const int* in_sizes, int n_in,
                              void* d_out, int out_size, void* d_ws,
                              size_t ws_size, hipStream_t stream) {
  const float* x  = (const float*)d_in[0];
  const int*   tp = (const int*)d_in[1];
  const float* Wq = (const float*)d_in[2];
  const float* Wk = (const float*)d_in[3];
  const float* Wv = (const float*)d_in[4];
  const float* Wo = (const float*)d_in[5];
  float* out = (float*)d_out;

  char* ws = (char*)d_ws;
  size_t off = 0;
  __bf16* xbf  = (__bf16*)(ws + off); off += (size_t)MTOT * D_MODEL * 2;
  __bf16* wqb  = (__bf16*)(ws + off); off += (size_t)D_MODEL * D_MODEL * 2;
  __bf16* wkb  = (__bf16*)(ws + off); off += (size_t)D_MODEL * D_MODEL * 2;
  __bf16* wvb  = (__bf16*)(ws + off); off += (size_t)D_MODEL * D_MODEL * 2;
  __bf16* wob  = (__bf16*)(ws + off); off += (size_t)D_MODEL * D_MODEL * 2;
  __bf16* qbuf = (__bf16*)(ws + off); off += (size_t)MTOT * D_MODEL * 2;
  __bf16* kbuf = (__bf16*)(ws + off); off += (size_t)MTOT * D_MODEL * 2;
  __bf16* vtb  = (__bf16*)(ws + off); off += (size_t)MTOT * D_MODEL * 2;
  __bf16* abuf = (__bf16*)(ws + off); off += (size_t)MTOT * D_MODEL * 2;
  (void)off; (void)in_sizes; (void)n_in; (void)out_size; (void)ws_size;

  {
    int n8 = MTOT * D_MODEL / 8;
    cvt_bf16_kernel<<<n8 / 256, 256, 0, stream>>>(x, xbf, n8);
    int w8 = D_MODEL * D_MODEL / 8;
    cvt_bf16_kernel<<<w8 / 256, 256, 0, stream>>>(Wq, wqb, w8);
    cvt_bf16_kernel<<<w8 / 256, 256, 0, stream>>>(Wk, wkb, w8);
    cvt_bf16_kernel<<<w8 / 256, 256, 0, stream>>>(Wv, wvb, w8);
    cvt_bf16_kernel<<<w8 / 256, 256, 0, stream>>>(Wo, wob, w8);
  }

  dim3 gg(MTOT / 64, D_MODEL / 128);
  gemm_wmma_kernel<0><<<gg, 128, 0, stream>>>(xbf, wqb, qbuf, nullptr);
  gemm_wmma_kernel<0><<<gg, 128, 0, stream>>>(xbf, wkb, kbuf, nullptr);
  gemm_wmma_kernel<1><<<gg, 128, 0, stream>>>(xbf, wvb, vtb, nullptr);

  int npairs = BATCH * N_HEADS * SEQ * (DK / 2);
  rope_kernel<<<npairs / 256, 256, 0, stream>>>(qbuf, tp, 0.125f, npairs);
  rope_kernel<<<npairs / 256, 256, 0, stream>>>(kbuf, tp, 1.0f, npairs);

  dim3 ga(SEQ / 64, BATCH * N_HEADS);
  flash_attn_kernel<<<ga, 128, 0, stream>>>(qbuf, kbuf, vtb, abuf);

  gemm_wmma_kernel<2><<<gg, 128, 0, stream>>>(abuf, wob, nullptr, out);
}
